// ConditionedGraphTransformer_69930657513932
// MI455X (gfx1250) — matrix operand order repaired and checked by reading, not compiled, not run on hardware
//
#include <hip/hip_runtime.h>
#include <hip/hip_bf16.h>

typedef _Float16 h8   __attribute__((ext_vector_type(8)));
typedef _Float16 v16h __attribute__((ext_vector_type(16)));
typedef float    v8f  __attribute__((ext_vector_type(8)));
typedef __attribute__((address_space(3))) _Float16 lds_f16;

#define HCONST 768
#define CDIM   256   // conditioned dim == out dim == 256 in this instance
#define ASTRIDE 72   // LDS A-panel row stride in halves (144B, odd # of 16B units)

// ---------------------------------------------------------------------------
__global__ void cvt_f16_kernel(const float* __restrict__ s, _Float16* __restrict__ d, long n) {
    long i = (long)blockIdx.x * blockDim.x + threadIdx.x;
    if (i < n) d[i] = (_Float16)s[i];
}

// W[In,Out] f32 -> Wt[Out,In] f16  (transposed so WMMA B-fragments are contiguous)
__global__ void wT_f16_kernel(const float* __restrict__ W, _Float16* __restrict__ Wt, int In, int Out) {
    int i = blockIdx.x * blockDim.x + threadIdx.x;
    if (i >= In * Out) return;
    int o = i % Out, r = i / Out;
    Wt[(long)o * In + r] = (_Float16)W[(long)r * Out + o];
}

// relation table input: rows 0..NREL-1 = edge_emb, row NREL = loop_rel, rest 0 (f16)
__global__ void build_relcat_kernel(const float* __restrict__ emb, const float* __restrict__ loop,
                                    _Float16* __restrict__ out, int NREL, int R) {
    int i = blockIdx.x * blockDim.x + threadIdx.x;
    if (i >= R * CDIM) return;
    int r = i / CDIM, c = i % CDIM;
    float v = (r < NREL) ? emb[(long)r * CDIM + c] : ((r == NREL) ? loop[c] : 0.f);
    out[i] = (_Float16)v;
}

// ---------------------------------------------------------------------------
// Big-M WMMA GEMM with async-to-LDS A staging.
// Block = 256 threads (8 waves) computes 64(M) x 128(N).
// Each wave: 16M x 64N = 4 accumulators; A fragment reused 4x from registers;
// A panel staged to LDS via global_load_async_to_lds_b128 (double-buffered);
// all 4 B fragments preloaded into distinct regs per k-step so loads pipeline.
__global__ __launch_bounds__(256)
void wmma_gemm_lds_kernel(const _Float16* __restrict__ A, const _Float16* __restrict__ Wt,
                          const float* __restrict__ bias, float* __restrict__ C,
                          int M, int K, int Nc, int ldc) {
    __shared__ _Float16 sA[2][64 * ASTRIDE];
    int nblk = Nc >> 7;
    int bm = blockIdx.x / nblk, bn = blockIdx.x % nblk;
    int tid = threadIdx.x, wave = tid >> 5, lane = tid & 31;
    int row0 = bm * 64;

    // stage one 64-row x 64-half K-panel (8KB): 512 x 16B units, 2 per thread
    auto stage = [&](int buf, int kb) {
        #pragma unroll
        for (int i = 0; i < 2; ++i) {
            int idx = tid + i * 256;              // 0..511
            int r = idx >> 3, u = idx & 7;        // row, 16B-unit within 128B row chunk
            unsigned ldsoff = (unsigned)(unsigned long)(lds_f16*)(&sA[buf][r * ASTRIDE + u * 8]);
            const _Float16* g = A + (long)(row0 + r) * K + kb + u * 8;
            asm volatile("global_load_async_to_lds_b128 %0, %1, off"
                         :: "v"(ldsoff), "v"(g) : "memory");
        }
    };

    int mtile  = wave >> 1;                       // 0..3
    int ntbase = bn * 128 + (wave & 1) * 64;
    int arow   = mtile * 16 + (lane & 15);
    int aoff   = (lane & 16) ? 16 : 0;            // A: lanes16-31 hold K 16..31
    int bcol15 = lane & 15;
    int boff   = (lane & 16) ? 16 : 0;            // B: lanes16-31 hold K 16..31

    const _Float16* brow0 = Wt + (long)(ntbase +  0 + bcol15) * K + boff;
    const _Float16* brow1 = Wt + (long)(ntbase + 16 + bcol15) * K + boff;
    const _Float16* brow2 = Wt + (long)(ntbase + 32 + bcol15) * K + boff;
    const _Float16* brow3 = Wt + (long)(ntbase + 48 + bcol15) * K + boff;

    v8f acc0 = {}, acc1 = {}, acc2 = {}, acc3 = {};
    int nkb = K >> 6;

    stage(0, 0);
    asm volatile("s_wait_asynccnt 0" ::: "memory");
    __syncthreads();

    for (int kbi = 0; kbi < nkb; ++kbi) {
        int buf = kbi & 1;
        if (kbi + 1 < nkb) stage(buf ^ 1, (kbi + 1) << 6);

        const _Float16* sa = &sA[buf][arow * ASTRIDE];
        #pragma unroll
        for (int kk = 0; kk < 64; kk += 32) {
            v16h a;
            ((h8*)&a)[0] = *(const h8*)(sa + kk + aoff);        // ds_load_b128
            ((h8*)&a)[1] = *(const h8*)(sa + kk + aoff + 8);
            int kg = (kbi << 6) + kk;
            v16h b0, b1, b2, b3;                                // 8 clause-able b128 loads
            ((h8*)&b0)[0] = *(const h8*)(brow0 + kg);
            ((h8*)&b0)[1] = *(const h8*)(brow0 + kg + 8);
            ((h8*)&b1)[0] = *(const h8*)(brow1 + kg);
            ((h8*)&b1)[1] = *(const h8*)(brow1 + kg + 8);
            ((h8*)&b2)[0] = *(const h8*)(brow2 + kg);
            ((h8*)&b2)[1] = *(const h8*)(brow2 + kg + 8);
            ((h8*)&b3)[0] = *(const h8*)(brow3 + kg);
            ((h8*)&b3)[1] = *(const h8*)(brow3 + kg + 8);
            acc0 = __builtin_amdgcn_wmma_f32_16x16x32_f16(false, a, false, b0, (short)0, acc0, false, false);
            acc1 = __builtin_amdgcn_wmma_f32_16x16x32_f16(false, a, false, b1, (short)0, acc1, false, false);
            acc2 = __builtin_amdgcn_wmma_f32_16x16x32_f16(false, a, false, b2, (short)0, acc2, false, false);
            acc3 = __builtin_amdgcn_wmma_f32_16x16x32_f16(false, a, false, b3, (short)0, acc3, false, false);
        }
        if (kbi + 1 < nkb) asm volatile("s_wait_asynccnt 0" ::: "memory");
        __syncthreads();
    }

    int mbase = row0 + mtile * 16 + ((lane & 16) ? 8 : 0);
    v8f accs[4] = {acc0, acc1, acc2, acc3};
    #pragma unroll
    for (int t = 0; t < 4; ++t) {
        int col = ntbase + t * 16 + bcol15;
        float bv = bias ? bias[col] : 0.f;
        #pragma unroll
        for (int r = 0; r < 8; ++r)
            C[(long)(mbase + r) * ldc + col] = accs[t][r] + bv;
    }
}

// Fallback WMMA GEMM (one wave per 16x16 tile) for small / odd shapes.
__global__ __launch_bounds__(128)
void wmma_gemm_kernel(const _Float16* __restrict__ A, const _Float16* __restrict__ Wt,
                      const float* __restrict__ bias, float* __restrict__ C,
                      int M, int K, int Nc, int ldc, int coloff) {
    int wave = blockIdx.x * (blockDim.x >> 5) + (threadIdx.x >> 5);
    int tilesN = Nc >> 4;
    int tiles  = (M >> 4) * tilesN;
    if (wave >= tiles) return;               // wave-uniform: EXEC stays all-1s for WMMA
    int lane = threadIdx.x & 31;
    int mt = wave / tilesN, nt = wave % tilesN;
    int arow = mt * 16 + (lane & 15);
    int bcol = nt * 16 + (lane & 15);
    int aoff = (lane & 16) ? 16 : 0;
    int boff = (lane & 16) ? 16 : 0;
    const _Float16* ap = A  + (long)arow * K + aoff;
    const _Float16* bp = Wt + (long)bcol * K + boff;
    v8f c = {};
    for (int kb = 0; kb < K; kb += 32) {
        v16h a, b;
        ((h8*)&a)[0] = *(const h8*)(ap + kb);
        ((h8*)&a)[1] = *(const h8*)(ap + kb + 8);
        ((h8*)&b)[0] = *(const h8*)(bp + kb);
        ((h8*)&b)[1] = *(const h8*)(bp + kb + 8);
        c = __builtin_amdgcn_wmma_f32_16x16x32_f16(false, a, false, b, (short)0, c, false, false);
    }
    float bv = bias ? bias[bcol] : 0.f;
    int mbase = mt * 16 + ((lane & 16) ? 8 : 0);
    #pragma unroll
    for (int r = 0; r < 8; ++r)
        C[(long)(mbase + r) * ldc + coloff + bcol] = c[r] + bv;
}

// ---------------------------------------------------------------------------
__global__ void concat_claim_kernel(const float* __restrict__ xcp, const int* __restrict__ batch,
                                    float* __restrict__ h, int N, int GH) {
    int i = blockIdx.x * blockDim.x + threadIdx.x;
    if (i >= N * GH) return;
    int n = i / GH, t = i % GH;
    h[(long)n * CDIM + GH + t] = xcp[(long)batch[n] * GH + t];
}

__device__ __forceinline__ unsigned f2sort(float f) {
    unsigned u = __float_as_uint(f);
    return (u & 0x80000000u) ? ~u : (u | 0x80000000u);
}
__device__ __forceinline__ float sort2f(unsigned u) {
    return __uint_as_float((u & 0x80000000u) ? (u & 0x7FFFFFFFu) : ~u);
}

// one wave per edge: score[e,h] = scale * <q[dst], k[src]+rel[et]>   (+ segment max)
__global__ void edge_scores_kernel(const float* __restrict__ q, const float* __restrict__ k,
                                   const float* __restrict__ rel, const int* __restrict__ ei,
                                   const int* __restrict__ et, int E, int N, int NREL,
                                   float* __restrict__ score, unsigned* __restrict__ mmax) {
    int e = blockIdx.x * (blockDim.x >> 5) + (threadIdx.x >> 5);
    if (e >= E + N) return;
    int lane = threadIdx.x & 31;
    int s, d, r;
    if (e < E) { s = ei[e]; d = ei[E + e]; r = et[e]; }
    else       { s = d = e - E; r = NREL; }
    int head = lane >> 3, sub = lane & 7;
    const float* qp = q   + (long)d * CDIM + head * 64;
    const float* kp = k   + (long)s * CDIM + head * 64;
    const float* rp = rel + (long)r * CDIM + head * 64;
    __builtin_prefetch(qp, 0, 1);          // global_prefetch_b8: random-gather rows
    __builtin_prefetch(kp, 0, 1);
    float acc = 0.f;
    #pragma unroll
    for (int j = 0; j < 8; ++j) { int dd = sub + j * 8; acc += qp[dd] * (kp[dd] + rp[dd]); }
    acc += __shfl_xor(acc, 1); acc += __shfl_xor(acc, 2); acc += __shfl_xor(acc, 4);
    if (sub == 0) {
        float sc = acc * 0.125f;                           // 64^-0.5
        score[(long)e * 4 + head] = sc;
        atomicMax(&mmax[(long)d * 4 + head], f2sort(sc));
    }
}

// ex = exp(score - m[dst]); den[dst] += ex   (thread per edge*head)
__global__ void edge_exp_kernel(const int* __restrict__ ei, int E, int N,
                                const unsigned* __restrict__ mmax,
                                float* __restrict__ score, float* __restrict__ den) {
    long i = (long)blockIdx.x * blockDim.x + threadIdx.x;
    if (i >= (long)(E + N) * 4) return;
    int e = (int)(i >> 2), h = (int)(i & 3);
    int d = (e < E) ? ei[E + e] : (e - E);
    float m  = sort2f(mmax[(long)d * 4 + h]);
    float ex = __expf(score[i] - m);
    score[i] = ex;
    atomicAdd(&den[(long)d * 4 + h], ex);
}

// hout[dst] += w[e,h] * (v[src] + rel[et])   (one wave per edge, 8 dims/lane)
__global__ void edge_agg_kernel(const float* __restrict__ v, const float* __restrict__ rel,
                                const int* __restrict__ ei, const int* __restrict__ et,
                                const float* __restrict__ score, const float* __restrict__ den,
                                float* __restrict__ hout, int E, int N, int NREL) {
    int e = blockIdx.x * (blockDim.x >> 5) + (threadIdx.x >> 5);
    if (e >= E + N) return;
    int lane = threadIdx.x & 31;
    int s, d, r;
    if (e < E) { s = ei[e]; d = ei[E + e]; r = et[e]; }
    else       { s = d = e - E; r = NREL; }
    float wv = 0.f;
    if (lane < 4) wv = score[(long)e * 4 + lane] / (den[(long)d * 4 + lane] + 1e-16f);
    const float* vp = v + (long)s * CDIM;
    const float* rp = rel + (long)r * CDIM;
    float* op = hout + (long)d * CDIM;
    __builtin_prefetch(vp, 0, 1);
    #pragma unroll
    for (int j = 0; j < 8; ++j) {
        int dd = lane + 32 * j;
        float w = __shfl(wv, dd >> 6);
        atomicAdd(&op[dd], w * (vp[dd] + rp[dd]));
    }
}

// h = [relu](LN(hout [+ hres]))  — one 256-thread block per node
__global__ void ln_kernel(const float* __restrict__ hout, const float* __restrict__ hres,
                          const float* __restrict__ g, const float* __restrict__ b,
                          float* __restrict__ out, int addRes, int doRelu) {
    __shared__ float sm[CDIM], sv[CDIM];
    int n = blockIdx.x, t = threadIdx.x;
    float x = hout[(long)n * CDIM + t];
    if (addRes) x += hres[(long)n * CDIM + t];
    sm[t] = x; sv[t] = x * x;
    __syncthreads();
    for (int st = CDIM / 2; st > 0; st >>= 1) {
        if (t < st) { sm[t] += sm[t + st]; sv[t] += sv[t + st]; }
        __syncthreads();
    }
    float mu  = sm[0] * (1.f / CDIM);
    float var = sv[0] * (1.f / CDIM) - mu * mu;
    float y = (x - mu) * rsqrtf(var + 1e-5f) * g[t] + b[t];
    if (doRelu) y = fmaxf(y, 0.f);
    out[(long)n * CDIM + t] = y;
}

// cross-attn: se[n] = exp(<Q[batch[n]], K[n]> * scale); bden[batch[n]] += se
__global__ void cross_scores_kernel(const float* __restrict__ Q, const float* __restrict__ K,
                                    const int* __restrict__ batch, int N,
                                    float* __restrict__ nse, float* __restrict__ bden) {
    int n = blockIdx.x * (blockDim.x >> 5) + (threadIdx.x >> 5);
    if (n >= N) return;
    int lane = threadIdx.x & 31;
    int bb = batch[n];
    const float* qp = Q + (long)bb * CDIM;
    const float* kp = K + (long)n * CDIM;
    float acc = 0.f;
    #pragma unroll
    for (int j = 0; j < 8; ++j) { int dd = lane + 32 * j; acc += qp[dd] * kp[dd]; }
    acc += __shfl_xor(acc, 16); acc += __shfl_xor(acc, 8);
    acc += __shfl_xor(acc, 4);  acc += __shfl_xor(acc, 2); acc += __shfl_xor(acc, 1);
    if (lane == 0) {
        float se = __expf(acc * 0.0625f);   // 256^-0.5
        nse[n] = se;
        atomicAdd(&bden[bb], se);
    }
}

__global__ void ctx_accum_kernel(const float* __restrict__ V, const float* __restrict__ nse,
                                 const float* __restrict__ bden, const int* __restrict__ batch,
                                 float* __restrict__ ctx, int N) {
    int n = blockIdx.x * (blockDim.x >> 5) + (threadIdx.x >> 5);
    if (n >= N) return;
    int lane = threadIdx.x & 31;
    int bb = batch[n];
    float w = nse[n] / (bden[bb] + 1e-8f);
    const float* vp = V + (long)n * CDIM;
    float* cp = ctx + (long)bb * CDIM;
    #pragma unroll
    for (int j = 0; j < 8; ++j) { int dd = lane + 32 * j; atomicAdd(&cp[dd], vp[dd] * w); }
}

// logits[b] = [ctx[b], claim[b]] . cls_w + cls_b     (one wave per graph)
__global__ void logits_kernel(const float* __restrict__ ctx, const float* __restrict__ claim,
                              const float* __restrict__ clsw, const float* __restrict__ clsb,
                              float* __restrict__ out) {
    int b = blockIdx.x, lane = threadIdx.x;
    float acc = 0.f;
    #pragma unroll
    for (int j = 0; j < 32; ++j) {
        int idx = lane + 32 * j;     // 0..1023
        float val = (idx < CDIM) ? ctx[(long)b * CDIM + idx]
                                 : claim[(long)b * HCONST + (idx - CDIM)];
        acc += val * clsw[idx];
    }
    acc += __shfl_xor(acc, 16); acc += __shfl_xor(acc, 8);
    acc += __shfl_xor(acc, 4);  acc += __shfl_xor(acc, 2); acc += __shfl_xor(acc, 1);
    if (lane == 0) out[b] = acc + clsb[0];
}

// ---------------------------------------------------------------------------
extern "C" void kernel_launch(void* const* d_in, const int* in_sizes, int n_in,
                              void* d_out, int out_size, void* d_ws, size_t ws_size,
                              hipStream_t stream) {
    const int H = HCONST;
    const int N    = in_sizes[0] / H;
    const int B    = in_sizes[1] / H;
    const int E    = in_sizes[2] / 2;
    const int GH   = in_sizes[5] / H;            // 128
    const int CD   = in_sizes[10];               // 256
    const int OUT  = in_sizes[21];               // 256
    const int NREL = in_sizes[9] / CD;           // 500
    const int NL   = in_sizes[11] / (CD * OUT);  // 2
    const int RREL = ((NREL + 1 + 15) / 16) * 16; // 512 (pad for WMMA)
    const int TE   = E + N;

    const float* x        = (const float*)d_in[0];
    const float* claim    = (const float*)d_in[1];
    const int*   ei       = (const int*)  d_in[2];
    const int*   etype    = (const int*)  d_in[3];
    const int*   batch    = (const int*)  d_in[4];
    const float* npw      = (const float*)d_in[5];
    const float* npb      = (const float*)d_in[6];
    const float* cpw      = (const float*)d_in[7];
    const float* cpb      = (const float*)d_in[8];
    const float* edge_emb = (const float*)d_in[9];
    const float* loop_rel = (const float*)d_in[10];
    const float* gqw = (const float*)d_in[11];  const float* gqb = (const float*)d_in[12];
    const float* gkw = (const float*)d_in[13];  const float* gkb = (const float*)d_in[14];
    const float* gvw = (const float*)d_in[15];  const float* gvb = (const float*)d_in[16];
    const float* gew = (const float*)d_in[17];  const float* geb = (const float*)d_in[18];
    const float* n0g = (const float*)d_in[19];  const float* n0b = (const float*)d_in[20];
    const float* lng = (const float*)d_in[21];  const float* lnb = (const float*)d_in[22];
    const float* aqw = (const float*)d_in[23];  const float* aqb = (const float*)d_in[24];
    const float* akw = (const float*)d_in[25];  const float* akb = (const float*)d_in[26];
    const float* avw = (const float*)d_in[27];  const float* avb = (const float*)d_in[28];
    const float* clsw = (const float*)d_in[29]; const float* clsb = (const float*)d_in[30];
    float* outp = (float*)d_out;

    // ---- workspace bump allocator
    char* wp = (char*)d_ws;
    auto alloc = [&](size_t bytes) { char* p = wp; wp += (bytes + 255) & ~(size_t)255; return p; };
    float*    h     = (float*)   alloc((size_t)N * CD * 4);
    float*    hout  = (float*)   alloc((size_t)N * CD * 4);
    float*    qb    = (float*)   alloc((size_t)N * CD * 4);
    float*    kb    = (float*)   alloc((size_t)N * CD * 4);
    float*    vb    = (float*)   alloc((size_t)N * CD * 4);
    _Float16* h_h   = (_Float16*)alloc((size_t)N * CD * 2);
    _Float16* claim_h = (_Float16*)alloc((size_t)B * H * 2);
    float*    score = (float*)   alloc((size_t)TE * 4 * 4);
    unsigned* mmax  = (unsigned*)alloc((size_t)N * 4 * 4);
    float*    den   = (float*)   alloc((size_t)N * 4 * 4);
    float*    relf  = (float*)   alloc((size_t)RREL * OUT * 4);
    _Float16* relcat= (_Float16*)alloc((size_t)RREL * CD * 2);
    float*    xcp   = (float*)   alloc((size_t)B * GH * 4);
    float*    Qb    = (float*)   alloc((size_t)B * OUT * 4);
    float*    nse   = (float*)   alloc((size_t)N * 4);
    float*    bden  = (float*)   alloc((size_t)B * 4);
    float*    ctx   = (float*)   alloc((size_t)B * OUT * 4);
    _Float16* npwT  = (_Float16*)alloc((size_t)H * GH * 2);
    _Float16* cpwT  = (_Float16*)alloc((size_t)H * GH * 2);
    _Float16* gwT   = (_Float16*)alloc((size_t)NL * 4 * CD * OUT * 2);
    _Float16* aqwT  = (_Float16*)alloc((size_t)H * OUT * 2);
    _Float16* akwT  = (_Float16*)alloc((size_t)OUT * OUT * 2);
    _Float16* avwT  = (_Float16*)alloc((size_t)OUT * OUT * 2);
    _Float16* x_h   = (_Float16*)qb;   // alias: x_h (N*H*2) fits in qb+kb, dead before layer 0

    auto grid1 = [](long n, int t) { return (unsigned)((n + t - 1) / t); };
    auto wT = [&](const float* W, _Float16* Wt, int In, int Out) {
        wT_f16_kernel<<<grid1((long)In * Out, 256), 256, 0, stream>>>(W, Wt, In, Out);
    };
    auto gemm = [&](const _Float16* A, const _Float16* Wt, const float* bias, float* C,
                    int M, int K, int Nc, int ldc, int coloff) {
        if ((M % 64 == 0) && (Nc % 128 == 0) && (K % 64 == 0) && coloff == 0) {
            int blocks = (M / 64) * (Nc / 128);
            wmma_gemm_lds_kernel<<<blocks, 256, 0, stream>>>(A, Wt, bias, C, M, K, Nc, ldc);
        } else {
            int tiles = (M / 16) * (Nc / 16);
            wmma_gemm_kernel<<<(tiles + 3) / 4, 128, 0, stream>>>(A, Wt, bias, C, M, K, Nc, ldc, coloff);
        }
    };

    // ---- weights -> f16 transposed
    wT(npw, npwT, H, GH);
    wT(cpw, cpwT, H, GH);
    for (int l = 0; l < NL; ++l) {
        wT(gqw + (long)l * CD * OUT, gwT + ((long)l * 4 + 0) * CD * OUT, CD, OUT);
        wT(gkw + (long)l * CD * OUT, gwT + ((long)l * 4 + 1) * CD * OUT, CD, OUT);
        wT(gvw + (long)l * CD * OUT, gwT + ((long)l * 4 + 2) * CD * OUT, CD, OUT);
        wT(gew + (long)l * CD * OUT, gwT + ((long)l * 4 + 3) * CD * OUT, CD, OUT);
    }
    wT(aqw, aqwT, H, OUT);
    wT(akw, akwT, OUT, OUT);
    wT(avw, avwT, OUT, OUT);

    // ---- activations -> f16 ; relation table input
    cvt_f16_kernel<<<grid1((long)N * H, 256), 256, 0, stream>>>(x, x_h, (long)N * H);
    cvt_f16_kernel<<<grid1((long)B * H, 256), 256, 0, stream>>>(claim, claim_h, (long)B * H);
    build_relcat_kernel<<<grid1((long)RREL * CD, 256), 256, 0, stream>>>(edge_emb, loop_rel, relcat, NREL, RREL);

    // ---- h = [x@Wn + bn , (claim@Wc + bc)[batch]]
    gemm(x_h, npwT, npb, h, N, H, GH, CD, 0);
    gemm(claim_h, cpwT, cpb, xcp, B, H, GH, GH, 0);
    concat_claim_kernel<<<grid1((long)N * GH, 256), 256, 0, stream>>>(xcp, batch, h, N, GH);

    // ---- RGAT layers
    for (int l = 0; l < NL; ++l) {
        cvt_f16_kernel<<<grid1((long)N * CD, 256), 256, 0, stream>>>(h, h_h, (long)N * CD);
        gemm(h_h, gwT + ((long)l * 4 + 0) * CD * OUT, gqb + (long)l * OUT, qb, N, CD, OUT, OUT, 0);
        gemm(h_h, gwT + ((long)l * 4 + 1) * CD * OUT, gkb + (long)l * OUT, kb, N, CD, OUT, OUT, 0);
        gemm(h_h, gwT + ((long)l * 4 + 2) * CD * OUT, gvb + (long)l * OUT, vb, N, CD, OUT, OUT, 0);
        gemm(relcat, gwT + ((long)l * 4 + 3) * CD * OUT, geb + (long)l * OUT, relf, RREL, CD, OUT, OUT, 0);

        hipMemsetAsync(mmax, 0, (size_t)N * 4 * 4, stream);
        hipMemsetAsync(den,  0, (size_t)N * 4 * 4, stream);
        hipMemsetAsync(hout, 0, (size_t)N * CD * 4, stream);

        edge_scores_kernel<<<grid1(TE, 8), 256, 0, stream>>>(qb, kb, relf, ei, etype, E, N, NREL, score, mmax);
        edge_exp_kernel<<<grid1((long)TE * 4, 256), 256, 0, stream>>>(ei, E, N, mmax, score, den);
        edge_agg_kernel<<<grid1(TE, 8), 256, 0, stream>>>(vb, relf, ei, etype, score, den, hout, E, N, NREL);

        int last = (l == NL - 1);
        ln_kernel<<<N, CDIM, 0, stream>>>(hout, h, last ? lng : n0g, last ? lnb : n0b,
                                          h, /*addRes=*/!last, /*doRelu=*/!last);
    }

    // ---- cross attention pool + classifier
    cvt_f16_kernel<<<grid1((long)N * CD, 256), 256, 0, stream>>>(h, h_h, (long)N * CD);
    gemm(h_h, akwT, akb, qb /*K*/, N, OUT, OUT, OUT, 0);
    gemm(h_h, avwT, avb, kb /*V*/, N, OUT, OUT, OUT, 0);
    gemm(claim_h, aqwT, aqb, Qb, B, H, OUT, OUT, 0);

    hipMemsetAsync(bden, 0, (size_t)B * 4, stream);
    hipMemsetAsync(ctx,  0, (size_t)B * OUT * 4, stream);
    cross_scores_kernel<<<grid1(N, 8), 256, 0, stream>>>(Qb, qb, batch, N, nse, bden);
    ctx_accum_kernel<<<grid1(N, 8), 256, 0, stream>>>(kb, nse, bden, batch, ctx, N);
    logits_kernel<<<B, 32, 0, stream>>>(ctx, claim, clsw, clsb, outp);
}
